// RecurrentMambaCell_53154515255857
// MI455X (gfx1250) — matrix operand order repaired and checked by reading, not compiled, not run on hardware
//
#include <hip/hip_runtime.h>
#include <hip/hip_bf16.h>

// Problem constants (from reference)
#define B_SZ    128
#define DMODEL  2048
#define DINNER  4096
#define NSTATE  128

typedef __attribute__((ext_vector_type(16))) __bf16 v16bf;
typedef __attribute__((ext_vector_type(8)))  float  v8f;

// ---------------------------------------------------------------------------
// Split-bf16 (3xBF16) WMMA GEMM:  Cout[m,n] = sum_k A[m,k]*Bm[n,k] + bias[n]
//   A   : row-major (M, Kd)
//   Bm  : row-major (N, Kd)   (i.e. the transposed operand as stored: W[n,k])
//   One wave computes one 16x16 tile; 4 waves per 128-thread block cover
//   4 consecutive N-tiles. Requires M%16==0, N%64==0, Kd%32==0 (true here).
// Fragment layouts follow CDNA5 ISA 7.12.2:
//   A (16x32 bf16): lane half = lane>>4 selects K-subblock {0,8};
//     elems j=0..7 -> K = 8*half + j ; j=8..15 -> K = 16 + 8*half + (j-8)
//   B (32x16 bf16): same K order as A, for column N = lane&15
//     elems j=0..7 -> K = 8*half + j ; j=8..15 -> K = 16 + 8*half + (j-8)
//   C/D (16x16 f32): VGPR r -> row M = r + 8*half, col N = lane&15
// ---------------------------------------------------------------------------
__global__ __launch_bounds__(128) void gemm_bf16x3(
    const float* __restrict__ A, const float* __restrict__ Bm,
    const float* __restrict__ bias, float* __restrict__ Cout,
    int M, int N, int Kd)
{
    const int lane = threadIdx.x & 31;
    const int wave = threadIdx.x >> 5;
    const int l15  = lane & 15;
    const int half = lane >> 4;

    const int m0 = blockIdx.y * 16;
    const int n0 = (blockIdx.x * 4 + wave) * 16;

    const float* arow = A  + (size_t)(m0 + l15) * Kd;
    const float* brow = Bm + (size_t)(n0 + l15) * Kd;
    const int kbA = half * 8;    // A K-subblock base within 16
    const int koB = half * 8;    // B K-subblock base within 16

    v8f acc = {};

    for (int k = 0; k < Kd; k += 32) {
        float af[16], bf[16];
        // A: 8 consecutive floats at k+kbA, 8 more at k+16+kbA
        {
            const float4* p0 = reinterpret_cast<const float4*>(arow + k + kbA);
            const float4* p1 = reinterpret_cast<const float4*>(arow + k + kbA + 16);
            float4 t;
            t = p0[0]; af[0]=t.x;  af[1]=t.y;  af[2]=t.z;  af[3]=t.w;
            t = p0[1]; af[4]=t.x;  af[5]=t.y;  af[6]=t.z;  af[7]=t.w;
            t = p1[0]; af[8]=t.x;  af[9]=t.y;  af[10]=t.z; af[11]=t.w;
            t = p1[1]; af[12]=t.x; af[13]=t.y; af[14]=t.z; af[15]=t.w;
        }
        // B: 8 consecutive floats at k+koB, 8 more at k+16+koB
        {
            const float4* p0 = reinterpret_cast<const float4*>(brow + k + koB);
            const float4* p1 = reinterpret_cast<const float4*>(brow + k + koB + 16);
            float4 t;
            t = p0[0]; bf[0]=t.x;  bf[1]=t.y;  bf[2]=t.z;  bf[3]=t.w;
            t = p0[1]; bf[4]=t.x;  bf[5]=t.y;  bf[6]=t.z;  bf[7]=t.w;
            t = p1[0]; bf[8]=t.x;  bf[9]=t.y;  bf[10]=t.z; bf[11]=t.w;
            t = p1[1]; bf[12]=t.x; bf[13]=t.y; bf[14]=t.z; bf[15]=t.w;
        }

        v16bf ah, al, bh, bl;
        #pragma unroll
        for (int j = 0; j < 16; ++j) {
            __bf16 h = (__bf16)af[j];
            ah[j] = h;
            al[j] = (__bf16)(af[j] - (float)h);
            __bf16 g = (__bf16)bf[j];
            bh[j] = g;
            bl[j] = (__bf16)(bf[j] - (float)g);
        }

        // 3xBF16 split accumulation in fp32 (a_lo*b_lo term negligible)
        acc = __builtin_amdgcn_wmma_f32_16x16x32_bf16(false, ah, false, bh,
                                                      (short)0, acc, false, false);
        acc = __builtin_amdgcn_wmma_f32_16x16x32_bf16(false, al, false, bh,
                                                      (short)0, acc, false, false);
        acc = __builtin_amdgcn_wmma_f32_16x16x32_bf16(false, ah, false, bl,
                                                      (short)0, acc, false, false);
    }

    const int col = n0 + l15;
    const float bb = bias[col];
    #pragma unroll
    for (int r = 0; r < 8; ++r) {
        const int m = m0 + half * 8 + r;
        Cout[(size_t)m * N + col] = acc[r] + bb;
    }
}

// ---------------------------------------------------------------------------
// dA = exp(A), 2 MB: computed once, reused x128 batches out of L2.
// ---------------------------------------------------------------------------
__global__ __launch_bounds__(256) void exp_kernel(const float* __restrict__ A,
                                                  float* __restrict__ dA)
{
    const int i = blockIdx.x * blockDim.x + threadIdx.x;
    dA[i] = __expf(A[i]);
}

// ---------------------------------------------------------------------------
// Depthwise conv (K=4) window + SiLU; also emits new_conv_state (window shift).
// ---------------------------------------------------------------------------
__global__ __launch_bounds__(256) void conv_silu(
    const float* __restrict__ xproj, const float* __restrict__ conv_state,
    const float* __restrict__ conv_w, float* __restrict__ out_conv,
    float* __restrict__ u_out)
{
    const int idx = blockIdx.x * blockDim.x + threadIdx.x;  // b*DINNER + d
    const int d   = idx & (DINNER - 1);

    const float c0 = conv_state[(size_t)idx * 3 + 0];
    const float c1 = conv_state[(size_t)idx * 3 + 1];
    const float c2 = conv_state[(size_t)idx * 3 + 2];
    const float xp = xproj[idx];
    const float4 w = *reinterpret_cast<const float4*>(conv_w + (size_t)d * 4);

    const float s  = c0 * w.x + c1 * w.y + c2 * w.z + xp * w.w;
    const float uu = s / (1.0f + __expf(-s));   // silu
    u_out[idx] = uu;

    float* oc = out_conv + (size_t)idx * 3;
    oc[0] = c1; oc[1] = c2; oc[2] = xp;
}

// ---------------------------------------------------------------------------
// SSM state update fused with the C-reduction and D-term:
//   new_ssm[b,n,d] = dA[n,d]*ssm[b,n,d] + B[n]*u[b,d]
//   y_inner[b,d]   = sum_n C[n]*new_ssm[b,n,d] + D[d]*u[b,d]
// Single streaming pass over the 268MB state (read+write) — HBM-bound.
// Threads map to consecutive d -> fully coalesced 128B/wave per access.
// ---------------------------------------------------------------------------
__global__ __launch_bounds__(256) void ssm_update(
    const float* __restrict__ ssm_state, const float* __restrict__ dA,
    const float* __restrict__ u, const float* __restrict__ Bp,
    const float* __restrict__ Cp, const float* __restrict__ Dp,
    float* __restrict__ new_ssm, float* __restrict__ y_inner)
{
    const int idx = blockIdx.x * blockDim.x + threadIdx.x;  // b*DINNER + d
    const int b   = idx >> 12;           // / DINNER
    const int d   = idx & (DINNER - 1);

    const float ub = u[idx];
    float acc = Dp[d] * ub;

    const float* ss = ssm_state + (size_t)b * NSTATE * DINNER + d;
    float*       od = new_ssm   + (size_t)b * NSTATE * DINNER + d;

    #pragma unroll 4
    for (int n = 0; n < NSTATE; ++n) {
        const int np = (n + 4 < NSTATE) ? (n + 4) : n;     // uniform, in-bounds
        __builtin_prefetch(ss + (size_t)np * DINNER, 0, 0); // global_prefetch_b8
        const float da = dA[n * DINNER + d];
        const float s  = ss[(size_t)n * DINNER];
        const float ns = fmaf(da, s, Bp[n] * ub);
        od[(size_t)n * DINNER] = ns;
        acc = fmaf(Cp[n], ns, acc);
    }
    y_inner[idx] = acc;
}

// ---------------------------------------------------------------------------
// Launch: GEMM1 -> exp(A) -> conv/SiLU -> SSM stream -> GEMM2
// ---------------------------------------------------------------------------
extern "C" void kernel_launch(void* const* d_in, const int* in_sizes, int n_in,
                              void* d_out, int out_size, void* d_ws, size_t ws_size,
                              hipStream_t stream)
{
    (void)in_sizes; (void)n_in; (void)out_size; (void)ws_size;

    const float* x_t        = (const float*)d_in[0];
    const float* conv_state = (const float*)d_in[1];
    const float* ssm_state  = (const float*)d_in[2];
    const float* W_in       = (const float*)d_in[3];   // (8192,2048); rows [0,4096) used
    const float* b_in       = (const float*)d_in[4];
    const float* conv_w     = (const float*)d_in[5];   // (4096,4)
    const float* A_p        = (const float*)d_in[6];   // (128,4096)
    const float* B_param    = (const float*)d_in[7];
    const float* C_param    = (const float*)d_in[8];
    const float* D_param    = (const float*)d_in[9];
    const float* W_out      = (const float*)d_in[10];  // (2048,4096)
    const float* b_out      = (const float*)d_in[11];

    float* out      = (float*)d_out;
    float* y_t      = out;                                   // 128*2048
    float* new_conv = out + (size_t)B_SZ * DMODEL;           // 128*4096*3
    float* new_ssm  = new_conv + (size_t)B_SZ * DINNER * 3;  // 128*128*4096

    float* ws       = (float*)d_ws;                          // 8 MB total
    float* xproj    = ws;                                    // 128*4096
    float* u        = xproj + (size_t)B_SZ * DINNER;         // 128*4096
    float* dA       = u     + (size_t)B_SZ * DINNER;         // 128*4096
    float* y_inner  = dA    + (size_t)NSTATE * DINNER;       // 128*4096

    // GEMM1: xproj = x_t @ W_in[:4096].T + b_in   (M=128, N=4096, K=2048)
    {
        dim3 grid(DINNER / 64, B_SZ / 16);
        gemm_bf16x3<<<grid, 128, 0, stream>>>(x_t, W_in, b_in, xproj,
                                              B_SZ, DINNER, DMODEL);
    }

    // dA = exp(A)
    exp_kernel<<<(NSTATE * DINNER) / 256, 256, 0, stream>>>(A_p, dA);

    // conv window + SiLU -> u, new_conv_state
    conv_silu<<<(B_SZ * DINNER) / 256, 256, 0, stream>>>(xproj, conv_state,
                                                         conv_w, new_conv, u);

    // SSM state stream (HBM-bound phase)
    ssm_update<<<(B_SZ * DINNER) / 256, 256, 0, stream>>>(
        ssm_state, dA, u, B_param, C_param, D_param, new_ssm, y_inner);

    // GEMM2: y_t = y_inner @ W_out.T + b_out   (M=128, N=2048, K=4096)
    {
        dim3 grid(DMODEL / 64, B_SZ / 16);
        gemm_bf16x3<<<grid, 128, 0, stream>>>(y_inner, W_out, b_out, y_t,
                                              B_SZ, DMODEL, DINNER);
    }
}
